// sLSTMBlockONNXCompatible_76974403879313
// MI455X (gfx1250) — matrix operand order repaired and checked by reading, not compiled, not run on hardware
//
#include <hip/hip_runtime.h>
#include <cstdint>
#include <cstddef>

#define B_ROWS 4096
#define DIM    2048
#define NHEAD  8
#define DHD    256
#define UPN    2730
#define UPP    2752   // 43 * 64

typedef __bf16 bf16_t;
typedef __attribute__((ext_vector_type(16))) __bf16 v16bf;
typedef __attribute__((ext_vector_type(8)))  __bf16 v8bf;
typedef __attribute__((ext_vector_type(8)))  float  v8f;
typedef int v4i_t __attribute__((__vector_size__(16)));

// ---------------- gfx1250 async global->LDS copy (guarded) ----------------
#ifndef TRY_ASYNC
#define TRY_ASYNC 1
#endif
#if TRY_ASYNC && defined(__has_builtin)
#if __has_builtin(__builtin_amdgcn_global_load_async_to_lds_b128) && \
    __has_builtin(__builtin_amdgcn_s_wait_asynccnt)
#define USE_ASYNC_LDS 1
#endif
#endif
#ifndef USE_ASYNC_LDS
#define USE_ASYNC_LDS 0
#endif

__device__ __forceinline__ void cp16_lds(bf16_t* ldst, const bf16_t* gsrc) {
#if USE_ASYNC_LDS
  __builtin_amdgcn_global_load_async_to_lds_b128((v4i_t*)gsrc, (v4i_t*)ldst, 0, 0);
#else
  *(v8bf*)ldst = *(const v8bf*)gsrc;
#endif
}
__device__ __forceinline__ void cp_wait_lds() {
#if USE_ASYNC_LDS
  __builtin_amdgcn_s_wait_asynccnt(0);
#endif
}

// ---------------- WMMA helpers (bf16 16x16x32, f32 accum) ----------------
__device__ __forceinline__ v8f wmma_bf16(v16bf a, v16bf b, v8f c) {
  return __builtin_amdgcn_wmma_f32_16x16x32_bf16(false, a, false, b, (short)0, c,
                                                 false, false);
}

// A fragment from an LDS tile [rows][ldb] (bf16). Lane L: row = base row + (L&15);
// elems 0..7 = K[half*8 .. +8), elems 8..15 = K[16+half*8 .. +8).
__device__ __forceinline__ v16bf a_frag_lds(const bf16_t* lds, int row, int ld,
                                            int half) {
  union { v16bf v; v8bf h[2]; } u;
  const bf16_t* r = lds + row * ld;
  u.h[0] = *(const v8bf*)(r + half * 8);
  u.h[1] = *(const v8bf*)(r + 16 + half * 8);
  return u.v;
}

// B fragment from a transposed (N-major) weight Wt[N][ldk] (bf16).
// Lane L: col = n0 + (L&15), elems 0..7 = K[k0+half*8 .. +8), elems 8..15 = K[k0+16+half*8 .. +8).
__device__ __forceinline__ v16bf b_frag_gl(const bf16_t* wt, int n0, int ldk,
                                           int k0, int lane) {
  union { v16bf v; v8bf h[2]; } u;
  const bf16_t* p = wt + (size_t)(n0 + (lane & 15)) * ldk + k0 + (lane >> 4) * 8;
  u.h[0] = *(const v8bf*)p;
  u.h[1] = *(const v8bf*)(p + 16);
  return u.v;
}

__device__ __forceinline__ void block_reduce2(float& s, float& s2) {
  for (int off = 16; off > 0; off >>= 1) {
    s += __shfl_xor(s, off);
    s2 += __shfl_xor(s2, off);
  }
  __shared__ float sh[16];
  int w = (int)threadIdx.x >> 5;
  if ((threadIdx.x & 31) == 0) { sh[w] = s; sh[w + 8] = s2; }
  __syncthreads();
  s = 0.f; s2 = 0.f;
  for (int i = 0; i < 8; ++i) { s += sh[i]; s2 += sh[i + 8]; }
  __syncthreads();
}

// ---------------- weight transpose + cast (f32[K][N] -> bf16[dN][dK]) ------
__global__ void transpose_cast_kernel(const float* __restrict__ src,
                                      bf16_t* __restrict__ dst, int K, int N,
                                      int dN, int dK, unsigned long long sHead,
                                      unsigned long long dHead) {
  int head = blockIdx.z;
  size_t idx = (size_t)blockIdx.x * blockDim.x + threadIdx.x;
  size_t total = (size_t)dN * dK;
  if (idx >= total) return;
  int n = (int)(idx / dK);
  int k = (int)(idx % dK);
  float v = (n < N && k < K) ? src[sHead * head + (size_t)k * N + n] : 0.f;
  dst[dHead * head + idx] = (bf16_t)v;
}

// ---------------- prep: LN -> conv -> SiLU; bf16 casts ---------------------
__global__ __launch_bounds__(256) void prep_kernel(
    const float* __restrict__ x, const float* __restrict__ h_prev,
    const float* __restrict__ ln_g, const float* __restrict__ ln_b,
    const float* __restrict__ conv_w, const float* __restrict__ conv_b,
    bf16_t* __restrict__ xb, bf16_t* __restrict__ xcb, bf16_t* __restrict__ hb) {
  __shared__ float xn[DIM];
  int b = blockIdx.x, t = threadIdx.x;
  const float* xr = x + (size_t)b * DIM;
  float v[8], s = 0.f, s2 = 0.f;
#pragma unroll
  for (int i = 0; i < 8; ++i) {
    float val = xr[t + 256 * i];
    v[i] = val; s += val; s2 += val * val;
  }
  block_reduce2(s, s2);
  float mu = s * (1.f / DIM);
  float var = s2 * (1.f / DIM) - mu * mu;
  float inv = rsqrtf(var + 1e-5f);
#pragma unroll
  for (int i = 0; i < 8; ++i) {
    int j = t + 256 * i;
    xn[j] = (v[i] - mu) * inv * ln_g[j] + ln_b[j];
  }
  __syncthreads();
  float w0 = conv_w[0], w1 = conv_w[1], w2 = conv_w[2], w3 = conv_w[3];
  float cb = conv_b[0];
#pragma unroll
  for (int i = 0; i < 8; ++i) {
    int j = t + 256 * i;
    float c = cb + w3 * xn[j];
    if (j >= 1) c += w2 * xn[j - 1];
    if (j >= 2) c += w1 * xn[j - 2];
    if (j >= 3) c += w0 * xn[j - 3];
    float sc = c / (1.f + expf(-c));  // SiLU
    size_t idx = (size_t)b * DIM + j;
    xcb[idx] = (bf16_t)sc;
    xb[idx] = (bf16_t)v[i];
    hb[idx] = (bf16_t)h_prev[idx];
  }
}

// ---------------- gate GEMMs: 4 waves = 4 gates, shared A tiles -----------
__global__ __launch_bounds__(128) void gates_gemm_kernel(
    const bf16_t* __restrict__ xb, const bf16_t* __restrict__ xcb,
    const bf16_t* __restrict__ hb, const bf16_t* __restrict__ wt,
    const bf16_t* __restrict__ rt, const float* bw0, const float* bw1,
    const float* bw2, const float* bw3, const float* br0, const float* br1,
    const float* br2, const float* br3, float* __restrict__ zp,
    float* __restrict__ ip, float* __restrict__ fp, float* __restrict__ op) {
  const int m0 = blockIdx.x * 64;
  const int nt = blockIdx.y;       // 0..3 (64-col tile within head)
  const int head = blockIdx.z;     // 0..7
  const int g = threadIdx.y;       // 0=z 1=i 2=f 3=o
  const int lane = threadIdx.x;
  const int tid = g * 32 + lane;

  const bf16_t* Wg = wt + ((size_t)g * NHEAD + head) * DHD * DHD;
  const bf16_t* Rg = rt + ((size_t)g * NHEAD + head) * DHD * DHD;

  __shared__ bf16_t sA[2][64][40];
  v8f acc[4][4] = {};

  for (int phase = 0; phase < 2; ++phase) {
    const bf16_t* src0 = phase ? hb : xb;
    const bf16_t* src1 = phase ? hb : xcb;
    const bf16_t* Bw = phase ? Rg : Wg;
    const int nchunk = phase ? 256 : 512;
    const int asel = phase ? 0 : ((g == 1 || g == 2) ? 1 : 0);
    for (int k0 = 0; k0 < DHD; k0 += 32) {
      __syncthreads();
      for (int c = tid; c < nchunk; c += 128) {
        int tile = c >> 8, cc = c & 255;
        int row = cc >> 2, ko = (cc & 3) * 8;
        const bf16_t* s = tile ? src1 : src0;
        cp16_lds(&sA[tile][row][ko],
                 s + (size_t)(m0 + row) * DIM + head * DHD + k0 + ko);
      }
      cp_wait_lds();
      __syncthreads();
      v16bf bfr[4];
#pragma unroll
      for (int ni = 0; ni < 4; ++ni)
        bfr[ni] = b_frag_gl(Bw, nt * 64 + ni * 16, DHD, k0, lane);
#pragma unroll
      for (int mi = 0; mi < 4; ++mi) {
        v16bf af = a_frag_lds(&sA[asel][0][0], mi * 16 + (lane & 15), 40, lane >> 4);
#pragma unroll
        for (int ni = 0; ni < 4; ++ni)
          acc[mi][ni] = wmma_bf16(af, bfr[ni], acc[mi][ni]);
      }
    }
  }

  float* outp = (g == 0) ? zp : (g == 1) ? ip : (g == 2) ? fp : op;
  const float* bw = (g == 0) ? bw0 : (g == 1) ? bw1 : (g == 2) ? bw2 : bw3;
  const float* br = (g == 0) ? br0 : (g == 1) ? br1 : (g == 2) ? br2 : br3;
  const int hf = lane >> 4, nl = lane & 15;
#pragma unroll
  for (int mi = 0; mi < 4; ++mi)
#pragma unroll
    for (int ni = 0; ni < 4; ++ni) {
      int col = head * DHD + nt * 64 + ni * 16 + nl;
      float bias = bw[col] + br[col];
#pragma unroll
      for (int v = 0; v < 8; ++v) {
        int row = m0 + mi * 16 + hf * 8 + v;
        outp[(size_t)row * DIM + col] = acc[mi][ni][v] + bias;
      }
    }
}

// ---------------- cell update + group LayerNorm ----------------------------
__global__ __launch_bounds__(256) void cell_kernel(
    const float* __restrict__ zp, const float* __restrict__ ip,
    const float* __restrict__ fp, const float* __restrict__ op,
    const float* __restrict__ c_prev, const float* __restrict__ n_prev,
    const float* __restrict__ m_prev, const float* __restrict__ gn_g,
    const float* __restrict__ gn_b, float* __restrict__ out,
    bf16_t* __restrict__ onrm) {
  const size_t SEC = (size_t)B_ROWS * DIM;
  int b = blockIdx.x, t = threadIdx.x;
  size_t base = (size_t)b * DIM;
  float hv[8], s = 0.f, s2 = 0.f;
#pragma unroll
  for (int i = 0; i < 8; ++i) {
    int j = t + 256 * i;
    size_t idx = base + j;
    float it = ip[idx], ft = fp[idx];
    float z = tanhf(zp[idx]);
    float o = 1.f / (1.f + expf(-op[idx]));
    float mp = m_prev[idx];
    float mt = fmaxf(ft + mp, it);
    float ii = expf(it - mt);
    float ff = expf(ft + mp - mt);
    float ct = ff * c_prev[idx] + ii * z;
    float nt = ff * n_prev[idx] + ii;
    float ht = o * ct / nt;
    out[SEC + idx] = ht;
    out[2 * SEC + idx] = ct;
    out[3 * SEC + idx] = nt;
    out[4 * SEC + idx] = mt;
    hv[i] = ht; s += ht; s2 += ht * ht;
  }
  block_reduce2(s, s2);
  float mu = s * (1.f / DIM);
  float var = s2 * (1.f / DIM) - mu * mu;
  float inv = rsqrtf(var + 1e-5f);
#pragma unroll
  for (int i = 0; i < 8; ++i) {
    int j = t + 256 * i;
    float on = (hv[i] - mu) * inv * gn_g[j] + gn_b[j];
    onrm[base + j] = (bf16_t)on;
  }
}

// ---------------- up projection (left/right) + exact GELU gate -------------
__global__ __launch_bounds__(128) void up_gemm_kernel(
    const bf16_t* __restrict__ A, const bf16_t* __restrict__ uplt,
    const bf16_t* __restrict__ uprt, const float* __restrict__ lb,
    const float* __restrict__ rb, bf16_t* __restrict__ gated) {
  const int m0 = blockIdx.x * 64;
  const int n0 = blockIdx.y * 64;
  const int w = threadIdx.y, lane = threadIdx.x;
  const int tid = w * 32 + lane;
  const int msub = w & 1;
  const bf16_t* Wt = (w < 2) ? uplt : uprt;

  __shared__ bf16_t sA[64][40];
  __shared__ float sRes[2][64][68];
  v8f acc[2][4] = {};

  for (int k0 = 0; k0 < DIM; k0 += 32) {
    __syncthreads();
    for (int c = tid; c < 256; c += 128) {
      int row = c >> 2, ko = (c & 3) * 8;
      cp16_lds(&sA[row][ko], A + (size_t)(m0 + row) * DIM + k0 + ko);
    }
    if (k0 + 32 < DIM)
      __builtin_prefetch(A + (size_t)(m0 + (tid >> 1)) * DIM + k0 + 32, 0, 1);
    cp_wait_lds();
    __syncthreads();
    v16bf bfr[4];
#pragma unroll
    for (int ni = 0; ni < 4; ++ni)
      bfr[ni] = b_frag_gl(Wt, n0 + ni * 16, DIM, k0, lane);
#pragma unroll
    for (int mi = 0; mi < 2; ++mi) {
      v16bf af =
          a_frag_lds(&sA[0][0], msub * 32 + mi * 16 + (lane & 15), 40, lane >> 4);
#pragma unroll
      for (int ni = 0; ni < 4; ++ni)
        acc[mi][ni] = wmma_bf16(af, bfr[ni], acc[mi][ni]);
    }
  }
  __syncthreads();
  const int lr = (w < 2) ? 0 : 1;
  const float* bias = (w < 2) ? lb : rb;
  const int hf = lane >> 4, nl = lane & 15;
#pragma unroll
  for (int mi = 0; mi < 2; ++mi)
#pragma unroll
    for (int ni = 0; ni < 4; ++ni) {
      int col = n0 + ni * 16 + nl;
      float bs = (col < UPN) ? bias[col] : 0.f;
#pragma unroll
      for (int v = 0; v < 8; ++v) {
        int r = msub * 32 + mi * 16 + hf * 8 + v;
        sRes[lr][r][ni * 16 + nl] = acc[mi][ni][v] + bs;
      }
    }
  __syncthreads();
  for (int c = tid; c < 4096; c += 128) {
    int r = c >> 6, cc = c & 63;
    float l = sRes[0][r][cc], rr = sRes[1][r][cc];
    float gl = 0.5f * rr * (1.f + erff(rr * 0.70710678118654752f));
    gated[(size_t)(m0 + r) * UPP + n0 + cc] = (bf16_t)(l * gl);
  }
}

// ---------------- down projection + bias + residual ------------------------
__global__ __launch_bounds__(128) void down_gemm_kernel(
    const bf16_t* __restrict__ gated, const bf16_t* __restrict__ downt,
    const float* __restrict__ db, const float* __restrict__ x,
    float* __restrict__ out) {
  const int m0 = blockIdx.x * 64;
  const int n0 = blockIdx.y * 256 + threadIdx.y * 64;
  const int lane = threadIdx.x;
  const int tid = threadIdx.y * 32 + lane;

  __shared__ bf16_t sA[64][40];
  v8f acc[4][4] = {};

  for (int k0 = 0; k0 < UPP; k0 += 32) {
    __syncthreads();
    for (int c = tid; c < 256; c += 128) {
      int row = c >> 2, ko = (c & 3) * 8;
      cp16_lds(&sA[row][ko], gated + (size_t)(m0 + row) * UPP + k0 + ko);
    }
    cp_wait_lds();
    __syncthreads();
    v16bf bfr[4];
#pragma unroll
    for (int ni = 0; ni < 4; ++ni)
      bfr[ni] = b_frag_gl(downt, n0 + ni * 16, UPP, k0, lane);
#pragma unroll
    for (int mi = 0; mi < 4; ++mi) {
      v16bf af = a_frag_lds(&sA[0][0], mi * 16 + (lane & 15), 40, lane >> 4);
#pragma unroll
      for (int ni = 0; ni < 4; ++ni)
        acc[mi][ni] = wmma_bf16(af, bfr[ni], acc[mi][ni]);
    }
  }
  const int hf = lane >> 4, nl = lane & 15;
#pragma unroll
  for (int mi = 0; mi < 4; ++mi)
#pragma unroll
    for (int ni = 0; ni < 4; ++ni) {
      int col = n0 + ni * 16 + nl;
      float bs = db[col];
#pragma unroll
      for (int v = 0; v < 8; ++v) {
        int r = m0 + mi * 16 + hf * 8 + v;
        out[(size_t)r * DIM + col] = acc[mi][ni][v] + bs + x[(size_t)r * DIM + col];
      }
    }
}

// ---------------- host orchestration --------------------------------------
extern "C" void kernel_launch(void* const* d_in, const int* in_sizes, int n_in,
                              void* d_out, int out_size, void* d_ws,
                              size_t ws_size, hipStream_t stream) {
  (void)in_sizes; (void)n_in; (void)out_size; (void)ws_size;
  const float* x      = (const float*)d_in[0];
  const float* h_prev = (const float*)d_in[1];
  const float* c_prev = (const float*)d_in[2];
  const float* n_prev = (const float*)d_in[3];
  const float* m_prev = (const float*)d_in[4];
  const float* ln_g   = (const float*)d_in[5];
  const float* ln_b   = (const float*)d_in[6];
  const float* conv_w = (const float*)d_in[7];
  const float* conv_b = (const float*)d_in[8];
  const float* gn_g   = (const float*)d_in[9];
  const float* gn_b   = (const float*)d_in[10];
  const float* upl_w  = (const float*)d_in[11];
  const float* upl_b  = (const float*)d_in[12];
  const float* upr_w  = (const float*)d_in[13];
  const float* upr_b  = (const float*)d_in[14];
  const float* down_w = (const float*)d_in[15];
  const float* down_b = (const float*)d_in[16];
  // gate order used internally: z, i, f, o
  const float* Ww[4] = {(const float*)d_in[17], (const float*)d_in[19],
                        (const float*)d_in[21], (const float*)d_in[23]};
  const float* Wb[4] = {(const float*)d_in[18], (const float*)d_in[20],
                        (const float*)d_in[22], (const float*)d_in[24]};
  const float* Rw[4] = {(const float*)d_in[25], (const float*)d_in[27],
                        (const float*)d_in[29], (const float*)d_in[31]};
  const float* Rb[4] = {(const float*)d_in[26], (const float*)d_in[28],
                        (const float*)d_in[30], (const float*)d_in[32]};

  char* p = (char*)d_ws;
  auto carve = [&](size_t bytes) {
    void* r = (void*)p;
    p += (bytes + 255) & ~(size_t)255;
    return r;
  };
  const size_t BD = (size_t)B_ROWS * DIM;
  bf16_t* xb    = (bf16_t*)carve(BD * 2);
  bf16_t* xcb   = (bf16_t*)carve(BD * 2);
  bf16_t* hb    = (bf16_t*)carve(BD * 2);
  bf16_t* onrm  = (bf16_t*)carve(BD * 2);
  bf16_t* gated = (bf16_t*)carve((size_t)B_ROWS * UPP * 2);
  bf16_t* wt    = (bf16_t*)carve((size_t)4 * NHEAD * DHD * DHD * 2);
  bf16_t* rt    = (bf16_t*)carve((size_t)4 * NHEAD * DHD * DHD * 2);
  bf16_t* uplt  = (bf16_t*)carve((size_t)UPP * DIM * 2);
  bf16_t* uprt  = (bf16_t*)carve((size_t)UPP * DIM * 2);
  bf16_t* downt = (bf16_t*)carve((size_t)DIM * UPP * 2);
  float*  zp    = (float*)carve(BD * 4);
  float*  ipre  = (float*)carve(BD * 4);
  float*  fpre  = (float*)carve(BD * 4);
  float*  opre  = (float*)carve(BD * 4);

  // ---- weight transposes (f32 [K][N] -> bf16 [dN][dK], zero-padded) ----
  {
    const unsigned long long hstride = (unsigned long long)DHD * DHD;
    dim3 g((DHD * DHD + 255) / 256, 1, NHEAD);
    for (int gi = 0; gi < 4; ++gi) {
      transpose_cast_kernel<<<g, 256, 0, stream>>>(
          Ww[gi], wt + (size_t)gi * NHEAD * DHD * DHD, DHD, DHD, DHD, DHD,
          hstride, hstride);
      transpose_cast_kernel<<<g, 256, 0, stream>>>(
          Rw[gi], rt + (size_t)gi * NHEAD * DHD * DHD, DHD, DHD, DHD, DHD,
          hstride, hstride);
    }
  }
  {
    size_t tot = (size_t)UPP * DIM;
    dim3 g((unsigned)((tot + 255) / 256), 1, 1);
    transpose_cast_kernel<<<g, 256, 0, stream>>>(upl_w, uplt, DIM, UPN, UPP, DIM,
                                                 0, 0);
    transpose_cast_kernel<<<g, 256, 0, stream>>>(upr_w, uprt, DIM, UPN, UPP, DIM,
                                                 0, 0);
    transpose_cast_kernel<<<g, 256, 0, stream>>>(down_w, downt, UPN, DIM, DIM,
                                                 UPP, 0, 0);
  }

  // ---- pipeline ----
  prep_kernel<<<B_ROWS, 256, 0, stream>>>(x, h_prev, ln_g, ln_b, conv_w, conv_b,
                                          xb, xcb, hb);

  gates_gemm_kernel<<<dim3(B_ROWS / 64, 4, NHEAD), dim3(32, 4), 0, stream>>>(
      xb, xcb, hb, wt, rt, Wb[0], Wb[1], Wb[2], Wb[3], Rb[0], Rb[1], Rb[2],
      Rb[3], zp, ipre, fpre, opre);

  cell_kernel<<<B_ROWS, 256, 0, stream>>>(zp, ipre, fpre, opre, c_prev, n_prev,
                                          m_prev, gn_g, gn_b, (float*)d_out,
                                          onrm);

  up_gemm_kernel<<<dim3(B_ROWS / 64, UPP / 64), dim3(32, 4), 0, stream>>>(
      onrm, uplt, uprt, upl_b, upr_b, gated);

  down_gemm_kernel<<<dim3(B_ROWS / 64, DIM / 256), dim3(32, 4), 0, stream>>>(
      gated, downt, down_b, x, (float*)d_out);
}
